// TitansBlockMAC_41635412968161
// MI455X (gfx1250) — compile-verified, matrix operand order checked by reading, not run on hardware
//
#include <hip/hip_runtime.h>
#include <math.h>
#include <stdint.h>

typedef long long ll;
typedef __attribute__((ext_vector_type(8)))  float  v8f;
typedef __attribute__((ext_vector_type(16))) __bf16 v16bf;
typedef __attribute__((ext_vector_type(8)))  __bf16 v8bf;
typedef __attribute__((ext_vector_type(4)))  __bf16 v4bf;

constexpr int Bb  = 4;
constexpr int Ss  = 4096;
constexpr int Dd  = 1024;
constexpr int Hh  = 16;
constexpr int DHh = 64;
constexpr int Pp  = 16;
constexpr int Cc  = 512;
constexpr int FFf = 4096;
constexpr int NCc = 8;
constexpr int PC  = Pp + Cc;

// ---- CDNA5 async global->LDS helpers (ASYNCcnt-tracked DMA) ----
__device__ __forceinline__ void async_cp16(unsigned lds_byte, const void* gaddr) {
  asm volatile("global_load_async_to_lds_b128 %0, %1, off"
               :: "v"(lds_byte), "v"(gaddr) : "memory");
}
__device__ __forceinline__ void wait_async_le6() {
  asm volatile("s_wait_asynccnt 0x6" ::: "memory");
}
__device__ __forceinline__ void wait_async_0() {
  asm volatile("s_wait_asynccnt 0x0" ::: "memory");
}

#define LDP 40
constexpr int BM = 128, BN = 64, BK = 32;

template<int TAM, int TBM>
__global__ __launch_bounds__(128) void gemm_wmma_kernel(
    const void* __restrict__ Av, const void* __restrict__ Bv,
    const float* __restrict__ Cin, const float* __restrict__ bias,
    void* __restrict__ DoutV,
    int M, int N, int K, int lda, int ldb, int ldc, int ldd,
    float alpha, float beta, int mode, int act, int outBf, int innerN,
    ll sAo, ll sAi, ll sBo, ll sBi, ll sCo, ll sCi, ll sDo, ll sDi)
{
  const int z = blockIdx.z;
  const int outer = z / innerN, inner = z % innerN;
  const ll offA = outer * sAo + inner * sAi;
  const ll offB = outer * sBo + inner * sBi;
  const ll offD = outer * sDo + inner * sDi;
  if (Cin) Cin += outer * sCo + inner * sCi;

  const float*  Af  = (const float*)Av  + offA;
  const __bf16* Ab  = (const __bf16*)Av + offA;
  const float*  Bf  = (const float*)Bv  + offB;
  const __bf16* Bbp = (const __bf16*)Bv + offB;

  const int tileM = blockIdx.y * BM;
  const int tileN = blockIdx.x * BN;

  __shared__ __bf16 sA[2][BM * LDP];
  __shared__ __bf16 sB[2][BN * LDP];

  const int tid  = threadIdx.x;
  const int lane = tid & 31;
  const int w    = tid >> 5;
  const int waveR = (w >> 1) * 64;
  const int waveC = (w & 1) * 32;
  const int half = lane >> 4;
  const int l16  = lane & 15;

  const bool fullM = (tileM + BM <= M);
  const bool fullN = (tileN + BN <= N);

  v8f acc[4][2];
#pragma unroll
  for (int mi = 0; mi < 4; ++mi)
#pragma unroll
    for (int ni = 0; ni < 2; ++ni)
#pragma unroll
      for (int v = 0; v < 8; ++v) acc[mi][ni][v] = 0.f;

  auto mma_step = [&](const __bf16* pA, const __bf16* pB) {
    v16bf af[4], bfr[2];
#pragma unroll
    for (int mi = 0; mi < 4; ++mi) {
      const __bf16* p = pA + (waveR + mi * 16 + l16) * LDP + half * 8;
      v8bf lo = *(const v8bf*)p;
      v8bf hi = *(const v8bf*)(p + 16);
#pragma unroll
      for (int e = 0; e < 8; ++e) { af[mi][e] = lo[e]; af[mi][e + 8] = hi[e]; }
    }
#pragma unroll
    for (int ni = 0; ni < 2; ++ni) {
      const __bf16* p = pB + (waveC + ni * 16 + l16) * LDP + half * 8;
      v8bf lo = *(const v8bf*)p;
      v8bf hi = *(const v8bf*)(p + 16);
#pragma unroll
      for (int e = 0; e < 8; ++e) { bfr[ni][e] = lo[e]; bfr[ni][e + 8] = hi[e]; }
    }
#pragma unroll
    for (int mi = 0; mi < 4; ++mi)
#pragma unroll
      for (int ni = 0; ni < 2; ++ni)
        acc[mi][ni] = __builtin_amdgcn_wmma_f32_16x16x32_bf16(
            false, af[mi], false, bfr[ni], (short)0, acc[mi][ni], false, false);
  };

  const bool pipeOK = (TAM == 2) && (TBM == 2) && fullM && fullN && (K % BK == 0);

  if (pipeOK) {
    const unsigned aBase = (unsigned)(size_t)&sA[0][0];
    const unsigned bBase = (unsigned)(size_t)&sB[0][0];
    const unsigned aStr  = (unsigned)(BM * LDP * 2);
    const unsigned bStr  = (unsigned)(BN * LDP * 2);
    const int rA = w * 32 + (lane >> 2);
    const int rB = w * 16 + (lane >> 2);
    const int ck = lane & 3;

    auto issue = [&](int k0, int buf) {
      const __bf16* Abase = Ab + (ll)tileM * lda + k0 + ck * 8;
#pragma unroll
      for (int i = 0; i < 4; ++i) {
        const int row = rA + i * 8;
        async_cp16(aBase + (unsigned)buf * aStr + (unsigned)(row * (LDP * 2) + ck * 16),
                   Abase + (ll)row * lda);
      }
      const __bf16* Bbase = Bbp + (ll)tileN * ldb + k0 + ck * 8;
#pragma unroll
      for (int i = 0; i < 2; ++i) {
        const int row = rB + i * 8;
        async_cp16(bBase + (unsigned)buf * bStr + (unsigned)(row * (LDP * 2) + ck * 16),
                   Bbase + (ll)row * ldb);
      }
    };

    issue(0, 0);
    const int nIter = K / BK;
    for (int it = 0; it < nIter; ++it) {
      const int cur = it & 1;
      if (it + 1 < nIter) { issue((it + 1) * BK, cur ^ 1); wait_async_le6(); }
      else                { wait_async_0(); }
      __syncthreads();
      mma_step(&sA[cur][0], &sB[cur][0]);
      __syncthreads();
    }
  } else {
    for (int k0 = 0; k0 < K; k0 += BK) {
      const bool fullK = (k0 + BK <= K);
      bool usedAsync = false;

      if (TAM == 0) {
        if (fullM && fullK) {
          for (int r = tid; r < BM * BK / 4; r += 128) {
            const int m = r >> 3, kq = (r & 7) * 4;
            const float4 f = *(const float4*)(Af + (ll)(tileM + m) * lda + (k0 + kq));
            v4bf h; h[0]=(__bf16)f.x; h[1]=(__bf16)f.y; h[2]=(__bf16)f.z; h[3]=(__bf16)f.w;
            *(v4bf*)(&sA[0][m * LDP + kq]) = h;
          }
        } else {
          for (int t = tid; t < BM * BK; t += 128) {
            const int m = t >> 5, k = t & 31;
            const int gm = tileM + m, gk = k0 + k;
            const float v = (gm < M && gk < K) ? Af[(ll)gm * lda + gk] : 0.f;
            sA[0][m * LDP + k] = (__bf16)v;
          }
        }
      } else if (TAM == 1) {
        if (fullM && fullK) {
          for (int r = tid; r < BM * BK / 4; r += 128) {
            const int k = r >> 5, mq = (r & 31) * 4;
            const float4 f = *(const float4*)(Af + (ll)(k0 + k) * lda + (tileM + mq));
            sA[0][(mq + 0) * LDP + k] = (__bf16)f.x;
            sA[0][(mq + 1) * LDP + k] = (__bf16)f.y;
            sA[0][(mq + 2) * LDP + k] = (__bf16)f.z;
            sA[0][(mq + 3) * LDP + k] = (__bf16)f.w;
          }
        } else {
          for (int t = tid; t < BM * BK; t += 128) {
            const int k = t >> 7, m = t & 127;
            const int gm = tileM + m, gk = k0 + k;
            const float v = (gm < M && gk < K) ? Af[(ll)gk * lda + gm] : 0.f;
            sA[0][m * LDP + k] = (__bf16)v;
          }
        }
      } else {
        for (int t = tid; t < BM * BK; t += 128) {
          const int m = t >> 5, k = t & 31;
          const int gm = tileM + m, gk = k0 + k;
          sA[0][m * LDP + k] = (gm < M && gk < K) ? Ab[(ll)gm * lda + gk] : (__bf16)0.f;
        }
      }

      if (TBM == 0) {
        if (fullN && fullK) {
          for (int r = tid; r < BN * BK / 4; r += 128) {
            const int k = r >> 4, nq = (r & 15) * 4;
            const float4 f = *(const float4*)(Bf + (ll)(k0 + k) * ldb + (tileN + nq));
            sB[0][(nq + 0) * LDP + k] = (__bf16)f.x;
            sB[0][(nq + 1) * LDP + k] = (__bf16)f.y;
            sB[0][(nq + 2) * LDP + k] = (__bf16)f.z;
            sB[0][(nq + 3) * LDP + k] = (__bf16)f.w;
          }
        } else {
          for (int t = tid; t < BN * BK; t += 128) {
            const int k = t >> 6, n = t & 63;
            const int gn = tileN + n, gk = k0 + k;
            const float v = (gn < N && gk < K) ? Bf[(ll)gk * ldb + gn] : 0.f;
            sB[0][n * LDP + k] = (__bf16)v;
          }
        }
      } else {
        if (fullN && fullK) {
          const int rB2 = w * 16 + (lane >> 2), ck = lane & 3;
          const unsigned bBase = (unsigned)(size_t)&sB[0][0];
          const __bf16* Bbase = Bbp + (ll)tileN * ldb + k0 + ck * 8;
#pragma unroll
          for (int i = 0; i < 2; ++i) {
            const int row = rB2 + i * 8;
            async_cp16(bBase + (unsigned)(row * (LDP * 2) + ck * 16),
                       Bbase + (ll)row * ldb);
          }
          usedAsync = true;
        } else {
          for (int t = tid; t < BN * BK; t += 128) {
            const int n = t >> 5, k = t & 31;
            const int gn = tileN + n, gk = k0 + k;
            sB[0][n * LDP + k] = (gn < N && gk < K) ? Bbp[(ll)gn * ldb + gk] : (__bf16)0.f;
          }
        }
      }

      if (usedAsync) wait_async_0();
      __syncthreads();
      mma_step(&sA[0][0], &sB[0][0]);
      __syncthreads();
    }
  }

#pragma unroll
  for (int mi = 0; mi < 4; ++mi) {
#pragma unroll
    for (int ni = 0; ni < 2; ++ni) {
      const int n = tileN + waveC + ni * 16 + l16;
#pragma unroll
      for (int v = 0; v < 8; ++v) {
        const int m = tileM + waveR + mi * 16 + half * 8 + v;
        if (m < M && n < N) {
          float val = alpha * acc[mi][ni][v];
          if (bias) val += bias[n];
          if (Cin) {
            const float cv = Cin[(ll)m * ldc + n];
            if (mode == 1) val *= cv; else val += beta * cv;
          }
          if (act == 1)      val = 1.f / (1.f + expf(-val));
          else if (act == 2) val = val / (1.f + expf(-val));
          const ll oi = offD + (ll)m * ldd + n;
          if (outBf) ((__bf16*)DoutV)[oi] = (__bf16)val;
          else       ((float*)DoutV)[oi]  = val;
        }
      }
    }
  }
}

__global__ __launch_bounds__(256) void transpose_to_bf16_kernel(
    const float* __restrict__ W, __bf16* __restrict__ Wt, int K, int N)
{
  __shared__ float t[32][33];
  const int tx = threadIdx.x & 31, ty = threadIdx.x >> 5;
  const int n0 = blockIdx.x * 32, k0 = blockIdx.y * 32;
#pragma unroll
  for (int i = 0; i < 4; ++i)
    t[ty + i * 8][tx] = W[(ll)(k0 + ty + i * 8) * N + (n0 + tx)];
  __syncthreads();
#pragma unroll
  for (int i = 0; i < 4; ++i)
    Wt[(ll)(n0 + ty + i * 8) * K + (k0 + tx)] = (__bf16)t[tx][ty + i * 8];
}

__global__ __launch_bounds__(256) void ln_kernel(
    const float* __restrict__ A, ll aB, const float* __restrict__ R, ll rB,
    const float* __restrict__ gm, const float* __restrict__ bt,
    __bf16* __restrict__ Out, int rowsPerBatch)
{
  const int r = blockIdx.x;
  const int b = r / rowsPerBatch, i = r % rowsPerBatch;
  const float* a = A + (ll)b * aB + (ll)i * Dd;
  const float* rr = R ? (R + (ll)b * rB + (ll)i * Dd) : nullptr;
  __bf16* o = Out + (ll)r * Dd;
  __shared__ float red[256];
  const int tid = threadIdx.x;
  float lv[4]; float s = 0.f;
#pragma unroll
  for (int j = 0; j < 4; ++j) {
    const int c = tid + j * 256;
    float v = a[c];
    if (rr) v += rr[c];
    lv[j] = v; s += v;
  }
  red[tid] = s; __syncthreads();
  for (int st = 128; st > 0; st >>= 1) { if (tid < st) red[tid] += red[tid + st]; __syncthreads(); }
  const float mean = red[0] * (1.f / 1024.f);
  __syncthreads();
  s = 0.f;
#pragma unroll
  for (int j = 0; j < 4; ++j) { const float d = lv[j] - mean; s += d * d; }
  red[tid] = s; __syncthreads();
  for (int st = 128; st > 0; st >>= 1) { if (tid < st) red[tid] += red[tid + st]; __syncthreads(); }
  const float inv = rsqrtf(red[0] * (1.f / 1024.f) + 1e-5f);
#pragma unroll
  for (int j = 0; j < 4; ++j) {
    const int c = tid + j * 256;
    o[c] = (__bf16)((lv[j] - mean) * inv * gm[c] + bt[c]);
  }
}

__global__ __launch_bounds__(256) void rownorm_kernel(float* __restrict__ X)
{
  float* p = X + (ll)blockIdx.x * Dd;
  __shared__ float red[256];
  const int tid = threadIdx.x;
  float lv[4]; float s = 0.f;
#pragma unroll
  for (int j = 0; j < 4; ++j) { lv[j] = p[tid + j * 256]; s += lv[j] * lv[j]; }
  red[tid] = s; __syncthreads();
  for (int st = 128; st > 0; st >>= 1) { if (tid < st) red[tid] += red[tid + st]; __syncthreads(); }
  const float inv = rsqrtf(red[0] + 1e-6f);
#pragma unroll
  for (int j = 0; j < 4; ++j) p[tid + j * 256] = lv[j] * inv;
}

__global__ void update_mem_kernel(float* __restrict__ M, float* __restrict__ St,
                                  const float* __restrict__ grad,
                                  const float* __restrict__ lr,
                                  const float* __restrict__ mom,
                                  const float* __restrict__ fg, ll n)
{
  const ll i = (ll)blockIdx.x * 256 + threadIdx.x;
  if (i >= n) return;
  const float theta = 1.f / (1.f + expf(-lr[0]));
  const float eta   = 1.f / (1.f + expf(-mom[0]));
  const float alf   = 1.f / (1.f + expf(-fg[0]));
  const float s = eta * St[i] - theta * grad[i];
  St[i] = s;
  M[i]  = (1.f - alf) * M[i] + s;
}

__global__ void buildk_kernel(const float* __restrict__ pmem,
                              const float* __restrict__ kh,
                              __bf16* __restrict__ out)
{
  const ll total = (ll)Bb * Hh * PC * DHh;
  const ll idx = (ll)blockIdx.x * 256 + threadIdx.x;
  if (idx >= total) return;
  const int d = (int)(idx & (DHh - 1));
  ll r = idx >> 6;
  const int t = (int)(r % PC); r /= PC;
  const int h = (int)(r % Hh);
  const int b = (int)(r / Hh);
  float v;
  if (t < Pp) v = pmem[((ll)t * Hh + h) * DHh + d];
  else        v = kh[((ll)b * Cc + (t - Pp)) * (Hh * DHh) + h * DHh + d];
  out[idx] = (__bf16)v;
}

__global__ void buildvT_kernel(const float* __restrict__ pmem,
                               const float* __restrict__ vh,
                               __bf16* __restrict__ out)   // (B,H,DH,PC)
{
  const ll total = (ll)Bb * Hh * DHh * PC;
  const ll idx = (ll)blockIdx.x * 256 + threadIdx.x;
  if (idx >= total) return;
  const int t = (int)(idx % PC);
  ll r = idx / PC;
  const int d = (int)(r % DHh); r /= DHh;
  const int h = (int)(r % Hh);
  const int b = (int)(r / Hh);
  float v;
  if (t < Pp) v = pmem[((ll)t * Hh + h) * DHh + d];
  else        v = vh[((ll)b * Cc + (t - Pp)) * (Hh * DHh) + h * DHh + d];
  out[idx] = (__bf16)v;
}

__global__ __launch_bounds__(256) void softmax_kernel(float* __restrict__ sc,
                                                      __bf16* __restrict__ scb)
{
  const int row  = blockIdx.x * 8 + (threadIdx.x >> 5);
  const int lane = threadIdx.x & 31;
  const int qi = row % Cc;
  float* p = sc + (ll)row * PC;
  __bf16* ob = scb + (ll)row * PC;
  float mx = -3.4e38f;
  for (int j = lane; j < PC; j += 32) {
    const bool vis = (j < Pp) || ((j - Pp) <= qi);
    mx = fmaxf(mx, vis ? p[j] : -1e9f);
  }
  for (int off = 16; off > 0; off >>= 1) mx = fmaxf(mx, __shfl_xor(mx, off, 32));
  float sum = 0.f;
  for (int j = lane; j < PC; j += 32) {
    const bool vis = (j < Pp) || ((j - Pp) <= qi);
    const float e = vis ? expf(p[j] - mx) : 0.f;
    p[j] = e; sum += e;
  }
  for (int off = 16; off > 0; off >>= 1) sum += __shfl_xor(sum, off, 32);
  const float inv = 1.f / sum;
  for (int j = lane; j < PC; j += 32) ob[j] = (__bf16)(p[j] * inv);
}

__global__ void combine_kernel(const float* __restrict__ xchunk,
                               const float* __restrict__ g,
                               const float* __restrict__ ao,
                               const float* __restrict__ mc,
                               float* __restrict__ o)
{
  const ll idx = (ll)blockIdx.x * 256 + threadIdx.x;
  if (idx >= (ll)Bb * Cc * Dd) return;
  const int d = (int)(idx & (Dd - 1));
  const ll r = idx >> 10;
  const int i = (int)(r & (Cc - 1));
  const int b = (int)(r >> 9);
  const float xc = xchunk[(ll)b * Ss * Dd + (ll)i * Dd + d];
  const float gv = g[idx];
  o[idx] = xc + gv * ao[idx] + (1.f - gv) * mc[idx];
}

__global__ void zero_kernel(float* __restrict__ p, ll n)
{
  const ll i = (ll)blockIdx.x * 256 + threadIdx.x;
  if (i < n) p[i] = 0.f;
}

static void launch_gemm(hipStream_t s, int am, int bm,
    const void* A, const void* B, const float* Cin, const float* bias,
    void* D, int outBf,
    int M, int N, int K, int lda, int ldb, int ldc, int ldd,
    float alpha, float beta, int mode, int act,
    int batches, int innerN,
    ll sAo, ll sAi, ll sBo, ll sBi, ll sCo, ll sCi, ll sDo, ll sDi)
{
  dim3 g((N + BN - 1) / BN, (M + BM - 1) / BM, batches);
  if (am == 2 && bm == 2)
    gemm_wmma_kernel<2,2><<<g, 128, 0, s>>>(A, B, Cin, bias, D, M, N, K,
        lda, ldb, ldc, ldd, alpha, beta, mode, act, outBf, innerN,
        sAo, sAi, sBo, sBi, sCo, sCi, sDo, sDi);
  else if (am == 0 && bm == 2)
    gemm_wmma_kernel<0,2><<<g, 128, 0, s>>>(A, B, Cin, bias, D, M, N, K,
        lda, ldb, ldc, ldd, alpha, beta, mode, act, outBf, innerN,
        sAo, sAi, sBo, sBi, sCo, sCi, sDo, sDi);
  else if (am == 1 && bm == 0)
    gemm_wmma_kernel<1,0><<<g, 128, 0, s>>>(A, B, Cin, bias, D, M, N, K,
        lda, ldb, ldc, ldd, alpha, beta, mode, act, outBf, innerN,
        sAo, sAi, sBo, sBi, sCo, sCi, sDo, sDi);
  else
    gemm_wmma_kernel<0,0><<<g, 128, 0, s>>>(A, B, Cin, bias, D, M, N, K,
        lda, ldb, ldc, ldd, alpha, beta, mode, act, outBf, innerN,
        sAo, sAi, sBo, sBi, sCo, sCi, sDo, sDi);
}

extern "C" void kernel_launch(void* const* d_in, const int* in_sizes, int n_in,
                              void* d_out, int out_size, void* d_ws, size_t ws_size,
                              hipStream_t stream)
{
  (void)in_sizes; (void)n_in; (void)out_size; (void)ws_size;
  const float* x     = (const float*)d_in[0];
  const float* g1    = (const float*)d_in[1];
  const float* b1    = (const float*)d_in[2];
  const float* g2    = (const float*)d_in[3];
  const float* b2    = (const float*)d_in[4];
  const float* g3    = (const float*)d_in[5];
  const float* b3    = (const float*)d_in[6];
  const float* Wqm   = (const float*)d_in[7];
  const float* Wkm   = (const float*)d_in[8];
  const float* Wvm   = (const float*)d_in[9];
  const float* lr    = (const float*)d_in[10];
  const float* mom   = (const float*)d_in[11];
  const float* fg    = (const float*)d_in[12];
  const float* Wq    = (const float*)d_in[13];
  const float* bq    = (const float*)d_in[14];
  const float* Wk    = (const float*)d_in[15];
  const float* bk    = (const float*)d_in[16];
  const float* Wv    = (const float*)d_in[17];
  const float* bv    = (const float*)d_in[18];
  const float* Wo    = (const float*)d_in[19];
  const float* bo    = (const float*)d_in[20];
  const float* pk    = (const float*)d_in[21];
  const float* pv    = (const float*)d_in[22];
  const float* Wg    = (const float*)d_in[23];
  const float* bg    = (const float*)d_in[24];
  const float* Wgate = (const float*)d_in[25];
  const float* bgate = (const float*)d_in[26];
  const float* Wup   = (const float*)d_in[27];
  const float* bup   = (const float*)d_in[28];
  const float* Wdown = (const float*)d_in[29];
  const float* bdown = (const float*)d_in[30];
  float* out = (float*)d_out;

  const ll SZ_MD  = (ll)Bb * Dd * Dd;
  const ll SZ_BCD = (ll)Bb * Cc * Dd;
  const ll SZ_KV  = (ll)Bb * Hh * PC * DHh;
  const ll SZ_SC  = (ll)Bb * Hh * Cc * PC;
  const ll SZ_FF  = (ll)Bb * Cc * FFf;
  char* base = (char*)d_ws;
  size_t off = 0;
  auto allocF = [&](ll n) { float* p = (float*)(base + off);
                            off = (off + (size_t)n * 4 + 15) & ~(size_t)15; return p; };
  auto allocH = [&](ll n) { __bf16* p = (__bf16*)(base + off);
                            off = (off + (size_t)n * 2 + 15) & ~(size_t)15; return p; };
  float* wM    = allocF(SZ_MD);
  float* wS    = allocF(SZ_MD);
  float* qm    = allocF(SZ_BCD);
  float* km    = allocF(SZ_BCD);
  float* vm    = allocF(SZ_BCD);
  float* mc    = allocF(SZ_BCD);
  float* diff  = allocF(SZ_BCD);
  float* grad  = allocF(SZ_MD);
  float* kh    = allocF(SZ_BCD);
  float* vh    = allocF(SZ_BCD);
  float* sc    = allocF(SZ_SC);
  float* ao    = allocF(SZ_BCD);
  float* gb    = allocF(SZ_BCD);
  float* ob    = allocF(SZ_BCD);
  float* gateb = allocF(SZ_FF);
  __bf16* h1b   = allocH(SZ_BCD);
  __bf16* h2b   = allocH(SZ_BCD);
  __bf16* h3b   = allocH(SZ_BCD);
  __bf16* qhb   = allocH(SZ_BCD);
  __bf16* kfb   = allocH(SZ_KV);
  __bf16* vfTb  = allocH(SZ_KV);
  __bf16* scb   = allocH(SZ_SC);
  __bf16* aobb  = allocH(SZ_BCD);
  __bf16* prodb = allocH(SZ_FF);
  __bf16* tWqm   = allocH((ll)Dd * Dd);
  __bf16* tWkm   = allocH((ll)Dd * Dd);
  __bf16* tWvm   = allocH((ll)Dd * Dd);
  __bf16* tWq    = allocH((ll)Dd * Dd);
  __bf16* tWk    = allocH((ll)Dd * Dd);
  __bf16* tWv    = allocH((ll)Dd * Dd);
  __bf16* tWo    = allocH((ll)Dd * Dd);
  __bf16* tWg    = allocH((ll)Dd * Dd);
  __bf16* tWgate = allocH((ll)Dd * FFf);
  __bf16* tWup   = allocH((ll)Dd * FFf);
  __bf16* tWdown = allocH((ll)FFf * Dd);

  {
    dim3 gDD(Dd / 32, Dd / 32);
    transpose_to_bf16_kernel<<<gDD, 256, 0, stream>>>(Wqm, tWqm, Dd, Dd);
    transpose_to_bf16_kernel<<<gDD, 256, 0, stream>>>(Wkm, tWkm, Dd, Dd);
    transpose_to_bf16_kernel<<<gDD, 256, 0, stream>>>(Wvm, tWvm, Dd, Dd);
    transpose_to_bf16_kernel<<<gDD, 256, 0, stream>>>(Wq,  tWq,  Dd, Dd);
    transpose_to_bf16_kernel<<<gDD, 256, 0, stream>>>(Wk,  tWk,  Dd, Dd);
    transpose_to_bf16_kernel<<<gDD, 256, 0, stream>>>(Wv,  tWv,  Dd, Dd);
    transpose_to_bf16_kernel<<<gDD, 256, 0, stream>>>(Wo,  tWo,  Dd, Dd);
    transpose_to_bf16_kernel<<<gDD, 256, 0, stream>>>(Wg,  tWg,  Dd, Dd);
    dim3 gDF(FFf / 32, Dd / 32);
    transpose_to_bf16_kernel<<<gDF, 256, 0, stream>>>(Wgate, tWgate, Dd, FFf);
    transpose_to_bf16_kernel<<<gDF, 256, 0, stream>>>(Wup,   tWup,   Dd, FFf);
    dim3 gFD(Dd / 32, FFf / 32);
    transpose_to_bf16_kernel<<<gFD, 256, 0, stream>>>(Wdown, tWdown, FFf, Dd);
  }

  zero_kernel<<<(unsigned)((2 * SZ_MD + 255) / 256), 256, 0, stream>>>(wM, 2 * SZ_MD);

  const int rowsBC = Bb * Cc;
  const ll sBCD = (ll)Cc * Dd;

  for (int c = 0; c < NCc; ++c) {
    const float* xc = x + (ll)c * Cc * Dd;

    ln_kernel<<<rowsBC, 256, 0, stream>>>(xc, (ll)Ss * Dd, nullptr, 0, g1, b1, h1b, Cc);

    launch_gemm(stream, 2, 2, h1b, tWqm, nullptr, nullptr, qm, 0, rowsBC, Dd, Dd,
                Dd, Dd, 0, Dd, 1.f, 0.f, 0, 0, 1, 1, 0,0,0,0,0,0,0,0);
    launch_gemm(stream, 2, 2, h1b, tWkm, nullptr, nullptr, km, 0, rowsBC, Dd, Dd,
                Dd, Dd, 0, Dd, 1.f, 0.f, 0, 0, 1, 1, 0,0,0,0,0,0,0,0);
    launch_gemm(stream, 2, 2, h1b, tWvm, nullptr, nullptr, vm, 0, rowsBC, Dd, Dd,
                Dd, Dd, 0, Dd, 1.f, 0.f, 0, 0, 1, 1, 0,0,0,0,0,0,0,0);
    rownorm_kernel<<<rowsBC, 256, 0, stream>>>(km);

    launch_gemm(stream, 0, 0, qm, wM, nullptr, nullptr, mc, 0, Cc, Dd, Dd,
                Dd, Dd, 0, Dd, 1.f, 0.f, 0, 0, Bb, 1,
                sBCD, 0, (ll)Dd * Dd, 0, 0, 0, sBCD, 0);
    launch_gemm(stream, 0, 0, km, wM, vm, nullptr, diff, 0, Cc, Dd, Dd,
                Dd, Dd, Dd, Dd, 1.f, -1.f, 0, 0, Bb, 1,
                sBCD, 0, (ll)Dd * Dd, 0, sBCD, 0, sBCD, 0);
    launch_gemm(stream, 1, 0, km, diff, nullptr, nullptr, grad, 0, Dd, Dd, Cc,
                Dd, Dd, 0, Dd, 1.f / (float)Cc, 0.f, 0, 0, Bb, 1,
                sBCD, 0, sBCD, 0, 0, 0, (ll)Dd * Dd, 0);
    update_mem_kernel<<<(unsigned)((SZ_MD + 255) / 256), 256, 0, stream>>>(
        wM, wS, grad, lr, mom, fg, SZ_MD);

    ln_kernel<<<rowsBC, 256, 0, stream>>>(xc, (ll)Ss * Dd, mc, sBCD, g2, b2, h2b, Cc);

    launch_gemm(stream, 2, 2, h2b, tWq, nullptr, bq, qhb, 1, rowsBC, Hh * DHh, Dd,
                Dd, Dd, 0, Hh * DHh, 1.f, 0.f, 0, 0, 1, 1, 0,0,0,0,0,0,0,0);
    launch_gemm(stream, 2, 2, h2b, tWk, nullptr, bk, kh, 0, rowsBC, Hh * DHh, Dd,
                Dd, Dd, 0, Hh * DHh, 1.f, 0.f, 0, 0, 1, 1, 0,0,0,0,0,0,0,0);
    launch_gemm(stream, 2, 2, h2b, tWv, nullptr, bv, vh, 0, rowsBC, Hh * DHh, Dd,
                Dd, Dd, 0, Hh * DHh, 1.f, 0.f, 0, 0, 1, 1, 0,0,0,0,0,0,0,0);
    buildk_kernel<<<(unsigned)((SZ_KV + 255) / 256), 256, 0, stream>>>(pk, kh, kfb);
    buildvT_kernel<<<(unsigned)((SZ_KV + 255) / 256), 256, 0, stream>>>(pv, vh, vfTb);

    launch_gemm(stream, 2, 2, qhb, kfb, nullptr, nullptr, sc, 0, Cc, PC, DHh,
                Hh * DHh, DHh, 0, PC, 0.125f, 0.f, 0, 0,
                Bb * Hh, Hh,
                (ll)Cc * Hh * DHh, (ll)DHh,
                (ll)Hh * PC * DHh, (ll)PC * DHh,
                0, 0,
                (ll)Hh * Cc * PC, (ll)Cc * PC);
    softmax_kernel<<<(Bb * Hh * Cc) / 8, 256, 0, stream>>>(sc, scb);
    launch_gemm(stream, 2, 2, scb, vfTb, nullptr, nullptr, aobb, 1, Cc, DHh, PC,
                PC, PC, 0, Hh * DHh, 1.f, 0.f, 0, 0,
                Bb * Hh, Hh,
                (ll)Hh * Cc * PC, (ll)Cc * PC,
                (ll)Hh * DHh * PC, (ll)DHh * PC,
                0, 0,
                (ll)Cc * Hh * DHh, (ll)DHh);
    launch_gemm(stream, 2, 2, aobb, tWo, nullptr, bo, ao, 0, rowsBC, Dd, Hh * DHh,
                Hh * DHh, Hh * DHh, 0, Dd, 1.f, 0.f, 0, 0, 1, 1, 0,0,0,0,0,0,0,0);
    launch_gemm(stream, 0, 2, mc, tWg, nullptr, bg, gb, 0, rowsBC, Dd, Dd,
                Dd, Dd, 0, Dd, 1.f, 0.f, 0, 1, 1, 1, 0,0,0,0,0,0,0,0);
    combine_kernel<<<(unsigned)((SZ_BCD + 255) / 256), 256, 0, stream>>>(xc, gb, ao, mc, ob);
    ln_kernel<<<rowsBC, 256, 0, stream>>>(ob, sBCD, nullptr, 0, g3, b3, h3b, Cc);
    launch_gemm(stream, 2, 2, h3b, tWgate, nullptr, bgate, gateb, 0, rowsBC, FFf, Dd,
                Dd, Dd, 0, FFf, 1.f, 0.f, 0, 2, 1, 1, 0,0,0,0,0,0,0,0);
    launch_gemm(stream, 2, 2, h3b, tWup, gateb, bup, prodb, 1, rowsBC, FFf, Dd,
                Dd, Dd, FFf, FFf, 1.f, 0.f, 1, 0, 1, 1, 0,0,0,0,0,0,0,0);
    launch_gemm(stream, 2, 2, prodb, tWdown, ob, bdown, out + (ll)c * Cc * Dd, 0,
                Cc, Dd, FFf, FFf, FFf, Dd, Dd, 1.f, 1.f, 0, 0,
                Bb, 1,
                (ll)Cc * FFf, 0, 0, 0, sBCD, 0, (ll)Ss * Dd, 0);
  }
}
